// CrossAttentionBlock_42193758716344
// MI455X (gfx1250) — matrix operand order repaired and checked by reading, not compiled, not run on hardware
//
#include <hip/hip_runtime.h>
#include <hip/hip_bf16.h>

// ---------------------------------------------------------------------------
// CDNA5 (gfx1250) cross-attention block, bf16 WMMA pipeline.
//   q = x @ Wq + bq            (big-tile GEMM, 65536x1024x1024)
//   k = p @ Wk + bk            (row-tile GEMM, 1232x1024x512)
//   v = p @ Wv + bv            (row-tile GEMM, 1232x1024x512)
//   attn = softmax(q k^T / 8) v  per (b, head)  (fused WMMA + TDM staging)
//   out = attn @ Ww + bw       (big-tile GEMM, f32 out)
// ---------------------------------------------------------------------------

typedef __attribute__((ext_vector_type(16))) __bf16 v16bf;
typedef __attribute__((ext_vector_type(8)))  __bf16 v8bf;
typedef __attribute__((ext_vector_type(8)))  float  v8f;
typedef __attribute__((ext_vector_type(4)))  unsigned int u32x4;
typedef __attribute__((ext_vector_type(8)))  int i32x8;
typedef __attribute__((ext_vector_type(4)))  int i32x4;

#if defined(__has_builtin)
#if __has_builtin(__builtin_amdgcn_tensor_load_to_lds)
#define ATHENA_HAVE_TDM 1
#endif
#endif

__device__ __forceinline__ v16bf combine16(v8bf lo, v8bf hi) {
  v16bf r;
#pragma unroll
  for (int i = 0; i < 8; ++i) { r[i] = lo[i]; r[8 + i] = hi[i]; }
  return r;
}

// Load a 16-element bf16 fragment as two 16B-aligned 8-element chunks.
__device__ __forceinline__ v16bf load_frag(const __bf16* p0, const __bf16* p1) {
  return combine16(*(const v8bf*)p0, *(const v8bf*)p1);
}

__device__ __forceinline__ v8f wmma_bf16(v16bf a, v16bf b, v8f c) {
  // D = A(16x32 bf16) * B(32x16 bf16) + C(16x16 f32)
  return __builtin_amdgcn_wmma_f32_16x16x32_bf16(
      false, a, false, b, (short)0, c, false, false);
}

__device__ __forceinline__ v8f zero_v8f() {
  v8f z;
#pragma unroll
  for (int i = 0; i < 8; ++i) z[i] = 0.0f;
  return z;
}

// ---------------------------------------------------------------------------
// Transpose + f32->bf16 convert: W (K x N) -> Wt (N x K) so WMMA B-operand
// columns (fixed N, contiguous K) are linear in memory.
// ---------------------------------------------------------------------------
__global__ void __launch_bounds__(256)
transpose_cvt(const float* __restrict__ W, __bf16* __restrict__ Wt,
              int K, int N) {
  __shared__ float tile[32][33];
  const int tx = threadIdx.x & 31;
  const int ty = threadIdx.x >> 5;
  const int bn = blockIdx.x * 32;
  const int bk = blockIdx.y * 32;
#pragma unroll
  for (int i = ty; i < 32; i += 8)
    tile[i][tx] = W[(size_t)(bk + i) * N + bn + tx];
  __syncthreads();
#pragma unroll
  for (int i = ty; i < 32; i += 8)
    Wt[(size_t)(bn + i) * K + bk + tx] = (__bf16)tile[tx][i];
}

#define GEMM_N 1024

// ---------------------------------------------------------------------------
// Big-tile GEMM: block tile 64(M) x 256(N), 8 waves as 2Mx4N, wave tile 32x64.
// A (f32 or bf16) and B (pre-transposed bf16, N-major) staged through LDS.
// Per-block weight traffic: 256 cols x K x 2B (0.5 MB at K=1024) vs 2 MB for
// a full-N row-tile block -> ~4x less L2 traffic on the two 137-GFLOP GEMMs.
// grid.x = N-blocks (4) so the blocks sharing an A stripe are L2-co-resident.
// ---------------------------------------------------------------------------
#define BT_AS 40   // LDS A row stride (bf16 elems), 16B-aligned, bank-staggered
#define BT_BS 40   // LDS B row stride

template <bool A_F32, bool OUT_BF16>
__global__ void __launch_bounds__(256)
gemm_bigtile(const void* __restrict__ Avoid, const __bf16* __restrict__ Wt,
             const float* __restrict__ bias, void* __restrict__ outv,
             int M, int K) {
  __shared__ __attribute__((aligned(16))) __bf16 lA[64 * BT_AS];
  __shared__ __attribute__((aligned(16))) __bf16 lB[256 * BT_BS];

  const int tid  = threadIdx.x;
  const int wave = tid >> 5;
  const int lane = tid & 31;
  const int lr   = lane & 15;
  const int half = lane >> 4;
  const int wm   = wave >> 2;                // 0..1 (M)
  const int wn   = wave & 3;                 // 0..3 (N)
  const int m0   = blockIdx.y * 64;
  const int n0   = blockIdx.x * 256;

  v8f acc[2][4];
#pragma unroll
  for (int mt = 0; mt < 2; ++mt)
#pragma unroll
    for (int nt = 0; nt < 4; ++nt) acc[mt][nt] = zero_v8f();

  // Staging: A 64x32 (thread t -> row t>>2, cols (t&3)*8..+8),
  //          B 256x32 (thread t -> col t, all 32 K elems).
  const int sar = tid >> 2, sac = (tid & 3) * 8;
  const int koff = half ? 16 : 0;
  const int alo  = half ? 16 : 0;

  for (int kb = 0; kb < K; kb += 32) {
    if (A_F32) {
      const float* src = (const float*)Avoid + (size_t)(m0 + sar) * K + kb + sac;
#pragma unroll
      for (int i = 0; i < 8; ++i)
        lA[sar * BT_AS + sac + i] = (__bf16)src[i];
    } else {
      const __bf16* src =
          (const __bf16*)Avoid + (size_t)(m0 + sar) * K + kb + sac;
      *(v8bf*)&lA[sar * BT_AS + sac] = *(const v8bf*)src;
    }
    {
      const __bf16* src = Wt + (size_t)(n0 + tid) * K + kb;
#pragma unroll
      for (int c = 0; c < 4; ++c)
        *(v8bf*)&lB[tid * BT_BS + c * 8] = *(const v8bf*)(src + c * 8);
    }
    __syncthreads();

    v16bf af[2];
#pragma unroll
    for (int mt = 0; mt < 2; ++mt) {
      const __bf16* ar = &lA[(wm * 32 + mt * 16 + lr) * BT_AS + alo];
      af[mt] = load_frag(ar, ar + 8);
    }
#pragma unroll
    for (int nt = 0; nt < 4; ++nt) {
      const __bf16* bp = &lB[(wn * 64 + nt * 16 + lr) * BT_BS + koff];
      v16bf bf = load_frag(bp, bp + 8);
#pragma unroll
      for (int mt = 0; mt < 2; ++mt)
        acc[mt][nt] = wmma_bf16(af[mt], bf, acc[mt][nt]);
    }
    __syncthreads();
  }

  // Epilogue: C layout — VGPR r = row (r + half*8), column = tile base + lr.
#pragma unroll
  for (int mt = 0; mt < 2; ++mt) {
    const int rowbase = m0 + wm * 32 + mt * 16 + (half ? 8 : 0);
#pragma unroll
    for (int nt = 0; nt < 4; ++nt) {
      const int col = n0 + wn * 64 + nt * 16 + lr;
      const float bn = bias[col];
      if (OUT_BF16) {
        __bf16* out = (__bf16*)outv;
#pragma unroll
        for (int r = 0; r < 8; ++r)
          out[(size_t)(rowbase + r) * GEMM_N + col] =
              (__bf16)(acc[mt][nt][r] + bn);
      } else {
        float* out = (float*)outv;
#pragma unroll
        for (int r = 0; r < 8; ++r)
          out[(size_t)(rowbase + r) * GEMM_N + col] = acc[mt][nt][r] + bn;
      }
    }
  }
}

// ---------------------------------------------------------------------------
// Row-tile GEMM (kept for the small K/V projections, M=1232 not 64-divisible).
// ---------------------------------------------------------------------------
#define A_STRIDE 40

template <bool A_F32, bool OUT_BF16>
__global__ void __launch_bounds__(256)
gemm_rowtile(const void* __restrict__ Avoid, const __bf16* __restrict__ Wt,
             const float* __restrict__ bias, void* __restrict__ outv,
             int M, int K) {
  __shared__ __attribute__((aligned(16))) __bf16 lds_a[16 * A_STRIDE];

  const int tid  = threadIdx.x;
  const int wave = tid >> 5;
  const int lane = tid & 31;
  const int lr   = lane & 15;
  const int half = lane >> 4;
  const int m0   = blockIdx.x * 16;
  const int n0   = wave * 128;

  v8f acc[8];
#pragma unroll
  for (int i = 0; i < 8; ++i) acc[i] = zero_v8f();

  const int koff = half ? 16 : 0;
  const __bf16* brow[8];
#pragma unroll
  for (int nt = 0; nt < 8; ++nt)
    brow[nt] = Wt + (size_t)(n0 + nt * 16 + lr) * K + koff;

  const int alo  = half ? 16 : 0;
  const int e    = tid * 2;
  const int arow = e >> 5;
  const int acol = e & 31;

  for (int kb = 0; kb < K; kb += 32) {
    if (A_F32) {
      const float* src = (const float*)Avoid + (size_t)(m0 + arow) * K + kb + acol;
      lds_a[arow * A_STRIDE + acol]     = (__bf16)src[0];
      lds_a[arow * A_STRIDE + acol + 1] = (__bf16)src[1];
    } else {
      const __bf16* src =
          (const __bf16*)Avoid + (size_t)(m0 + arow) * K + kb + acol;
      lds_a[arow * A_STRIDE + acol]     = src[0];
      lds_a[arow * A_STRIDE + acol + 1] = src[1];
    }
    __syncthreads();
    const __bf16* ar = &lds_a[lr * A_STRIDE + alo];
    v16bf afrag = load_frag(ar, ar + 8);
#pragma unroll
    for (int nt = 0; nt < 8; ++nt) {
      const __bf16* bp = brow[nt] + kb;
      v16bf bfrag = load_frag(bp, bp + 8);
      acc[nt] = wmma_bf16(afrag, bfrag, acc[nt]);
    }
    __syncthreads();
  }

  const int rowbase = m0 + (half ? 8 : 0);
#pragma unroll
  for (int nt = 0; nt < 8; ++nt) {
    const int col = n0 + nt * 16 + lr;
    const float bn = bias[col];
    if (OUT_BF16) {
      __bf16* out = (__bf16*)outv;
#pragma unroll
      for (int r = 0; r < 8; ++r)
        out[(size_t)(rowbase + r) * GEMM_N + col] = (__bf16)(acc[nt][r] + bn);
    } else {
      float* out = (float*)outv;
#pragma unroll
      for (int r = 0; r < 8; ++r)
        out[(size_t)(rowbase + r) * GEMM_N + col] = acc[nt][r] + bn;
    }
  }
}

// ---------------------------------------------------------------------------
// Fused attention. K tile staged by the Tensor Data Mover when available:
// D# 2D tile (tile 64x80, tensor_dim1=77 -> rows 77..79 auto-zero), LDS pad
// 12 DWORDs after every 32 DWORDs -> 176B row stride (= KT_STRIDE bf16).
// ---------------------------------------------------------------------------
#define KT_STRIDE 88   // 176B rows: 16B-aligned, bank-staggered (matches TDM pad)
#define VT_STRIDE 104
#define P_STRIDE  104

__global__ void __launch_bounds__(256)
attn_wmma(const __bf16* __restrict__ q, const __bf16* __restrict__ kmat,
          const __bf16* __restrict__ vmat, __bf16* __restrict__ attn_out) {
  __shared__ __attribute__((aligned(16))) __bf16 kt[80 * KT_STRIDE];
  __shared__ __attribute__((aligned(16))) __bf16 vt[64 * VT_STRIDE];
  __shared__ __attribute__((aligned(16))) __bf16 pl[8 * 16 * P_STRIDE];

  const int tid  = threadIdx.x;
  const int wave = tid >> 5;
  const int lane = tid & 31;
  const int lr   = lane & 15;
  const int half = lane >> 4;
  const int bh   = blockIdx.y;
  const int b    = bh >> 4;
  const int h    = bh & 15;
  const int q0   = blockIdx.x * 128 + wave * 16;

  const size_t kvbase = (size_t)b * 77 * 1024 + (size_t)h * 64;

#ifdef ATHENA_HAVE_TDM
  // TDM: DMA the 77x64 bf16 K tile (padded to 80 rows with zeros via OOB)
  // into kt with hardware LDS padding. Issued by wave 0 only.
  if (wave == 0) {
    const unsigned long long ga =
        (unsigned long long)(size_t)(const void*)(kmat + kvbase);
    const unsigned int ldsa = (unsigned int)(size_t)(void*)&kt[0];
    u32x4 g0;
    g0[0] = 1u;                                   // count=1 valid descriptor
    g0[1] = ldsa;                                 // lds_addr
    g0[2] = (unsigned int)ga;                     // global_addr[31:0]
    g0[3] = (unsigned int)((ga >> 32) & 0x01ffffffu) | (2u << 30); // type=2
    i32x8 g1;
    g1[0] = (1 << 16)    // data_size = 2B
          | (1 << 20)    // pad_enable
          | (4 << 22)    // pad_interval: every 32 DWORDs (= one 64-elem row)
          | (11 << 25);  // pad_amount: 12 DWORDs -> 176B row stride
    g1[1] = 64 << 16;    // tensor_dim0 = 64 (low 16 bits)
    g1[2] = 77 << 16;    // tensor_dim0 hi=0 | tensor_dim1 = 77 (low 16)
    g1[3] = 64 << 16;    // tensor_dim1 hi=0 | tile_dim0 = 64
    g1[4] = 80;          // tile_dim1 = 80 | tile_dim2 = 0
    g1[5] = 1024;        // tensor_dim0_stride = 1024 elems (low 32)
    g1[6] = 0;
    g1[7] = 0;
    i32x4 gz;
    gz[0] = gz[1] = gz[2] = gz[3] = 0;
#if __clang_major__ >= 23
    i32x8 g5;
#pragma unroll
    for (int i = 0; i < 8; ++i) g5[i] = 0;
    __builtin_amdgcn_tensor_load_to_lds(g0, g1, gz, gz, g5, 0);
#else
    __builtin_amdgcn_tensor_load_to_lds(g0, g1, gz, gz, 0);
#endif
#if __has_builtin(__builtin_amdgcn_s_wait_tensorcnt)
    __builtin_amdgcn_s_wait_tensorcnt(0);
#else
    asm volatile("s_wait_tensorcnt 0x0" ::: "memory");
#endif
  }
#else
  for (int i = tid; i < 80 * 64; i += 256) {
    const int key = i >> 6, d = i & 63;
    __bf16 val = (__bf16)0.0f;
    if (key < 77) val = kmat[kvbase + (size_t)key * 1024 + d];
    kt[key * KT_STRIDE + d] = val;
  }
#endif

  // Stage v transposed: [d][key], keys 77..95 zeroed (PV K-pad).
  for (int i = tid; i < 64 * 96; i += 256) {
    const int d = i / 96, key = i % 96;
    __bf16 val = (__bf16)0.0f;
    if (key < 77) val = vmat[kvbase + (size_t)key * 1024 + d];
    vt[d * VT_STRIDE + key] = val;
  }
  __syncthreads();

  // Q fragments over head dim (K=64 -> 2 chunks of 32).
  const int alo = half ? 16 : 0;
  const __bf16* qrow = q + ((size_t)b * 4096 + q0 + lr) * 1024 + (size_t)h * 64;
  v16bf aq0 = load_frag(qrow + alo,      qrow + alo + 8);
  v16bf aq1 = load_frag(qrow + 32 + alo, qrow + 32 + alo + 8);

  // scores: 5 key tiles of 16.
  const int bkoff = half ? 16 : 0;
  v8f s[5];
#pragma unroll
  for (int t = 0; t < 5; ++t) {
    const __bf16* kp = &kt[(t * 16 + lr) * KT_STRIDE + bkoff];
    v16bf bk0 = load_frag(kp,      kp + 8);
    v16bf bk1 = load_frag(kp + 32, kp + 40);
    v8f z = zero_v8f();
    z = wmma_bf16(aq0, bk0, z);
    s[t] = wmma_bf16(aq1, bk1, z);
  }

  // Softmax: each 16-lane half holds one row's 80 columns per VGPR slot.
  const float scale = 0.125f;                // 1/sqrt(64)
  __bf16* pw = &pl[wave * 16 * P_STRIDE];    // wave-private P tile
  {  // zero K-pad columns 80..95 of the P tile
    __bf16* zp = pw + lr * P_STRIDE + 80 + (half ? 8 : 0);
#pragma unroll
    for (int i = 0; i < 8; ++i) zp[i] = (__bf16)0.0f;
  }
#pragma unroll
  for (int r = 0; r < 8; ++r) {
    float sv[5];
    float m = -3.0e38f;
#pragma unroll
    for (int t = 0; t < 5; ++t) {
      const int j = t * 16 + lr;
      float v = s[t][r] * scale;
      if (j >= 77) v = -3.0e38f;
      sv[t] = v;
      m = fmaxf(m, v);
    }
#pragma unroll
    for (int msk = 1; msk < 16; msk <<= 1)
      m = fmaxf(m, __shfl_xor(m, msk, 32));
    float sum = 0.0f;
#pragma unroll
    for (int t = 0; t < 5; ++t) {
      const int j = t * 16 + lr;
      const float e = (j < 77) ? __expf(sv[t] - m) : 0.0f;
      sv[t] = e;
      sum += e;
    }
#pragma unroll
    for (int msk = 1; msk < 16; msk <<= 1)
      sum += __shfl_xor(sum, msk, 32);
    const float inv = 1.0f / sum;
    const int prow = r + (half ? 8 : 0);
#pragma unroll
    for (int t = 0; t < 5; ++t)
      pw[prow * P_STRIDE + t * 16 + lr] = (__bf16)(sv[t] * inv);
  }

  // out = P(16x96) @ V(96x64): relayout P via wave-private LDS tile.
  v8f o[4];
#pragma unroll
  for (int nt = 0; nt < 4; ++nt) o[nt] = zero_v8f();
#pragma unroll
  for (int kc = 0; kc < 3; ++kc) {
    const __bf16* pr = pw + lr * P_STRIDE + kc * 32 + alo;
    v16bf ap = load_frag(pr, pr + 8);
#pragma unroll
    for (int nt = 0; nt < 4; ++nt) {
      const __bf16* vp = &vt[(nt * 16 + lr) * VT_STRIDE + kc * 32 + bkoff];
      v16bf bv = load_frag(vp, vp + 8);
      o[nt] = wmma_bf16(ap, bv, o[nt]);
    }
  }

  // Store attn_out bf16 at (b*4096 + row)*1024 + h*64 + d.
  const size_t obase =
      ((size_t)b * 4096 + q0 + (half ? 8 : 0)) * 1024 + (size_t)h * 64;
#pragma unroll
  for (int nt = 0; nt < 4; ++nt) {
    const int d = nt * 16 + lr;
#pragma unroll
    for (int r = 0; r < 8; ++r)
      attn_out[obase + (size_t)r * 1024 + d] = (__bf16)o[nt][r];
  }
}

// ---------------------------------------------------------------------------
// Launch
// ---------------------------------------------------------------------------
extern "C" void kernel_launch(void* const* d_in, const int* in_sizes, int n_in,
                              void* d_out, int out_size, void* d_ws,
                              size_t ws_size, hipStream_t stream) {
  const float* x  = (const float*)d_in[0];   // (16, 4096, 1024)
  const float* p  = (const float*)d_in[1];   // (16, 77, 512)
  const float* Wq = (const float*)d_in[2];   // (1024, 1024)
  const float* bq = (const float*)d_in[3];
  const float* Wk = (const float*)d_in[4];   // (512, 1024)
  const float* bk = (const float*)d_in[5];
  const float* Wv = (const float*)d_in[6];   // (512, 1024)
  const float* bv = (const float*)d_in[7];
  const float* Ww = (const float*)d_in[8];   // (1024, 1024)
  const float* bw = (const float*)d_in[9];
  float* out = (float*)d_out;

  char* ws = (char*)d_ws;
  __bf16* WqT  = (__bf16*)ws; ws += (size_t)1024 * 1024 * 2;
  __bf16* WkT  = (__bf16*)ws; ws += (size_t)1024 * 512 * 2;
  __bf16* WvT  = (__bf16*)ws; ws += (size_t)1024 * 512 * 2;
  __bf16* WwT  = (__bf16*)ws; ws += (size_t)1024 * 1024 * 2;
  __bf16* qbuf = (__bf16*)ws; ws += (size_t)65536 * 1024 * 2;
  __bf16* kbuf = (__bf16*)ws; ws += (size_t)1232 * 1024 * 2;
  __bf16* vbuf = (__bf16*)ws; ws += (size_t)1232 * 1024 * 2;
  __bf16* obuf = (__bf16*)ws;                // attn output, 65536 x 1024 bf16

  transpose_cvt<<<dim3(32, 32), 256, 0, stream>>>(Wq, WqT, 1024, 1024);
  transpose_cvt<<<dim3(32, 16), 256, 0, stream>>>(Wk, WkT, 512, 1024);
  transpose_cvt<<<dim3(32, 16), 256, 0, stream>>>(Wv, WvT, 512, 1024);
  transpose_cvt<<<dim3(32, 32), 256, 0, stream>>>(Ww, WwT, 1024, 1024);

  gemm_bigtile<true, true><<<dim3(4, 1024), 256, 0, stream>>>(x, WqT, bq, qbuf,
                                                              65536, 1024);
  gemm_rowtile<true, true><<<1232 / 16, 256, 0, stream>>>(p, WkT, bk, kbuf,
                                                          1232, 512);
  gemm_rowtile<true, true><<<1232 / 16, 256, 0, stream>>>(p, WvT, bv, vbuf,
                                                          1232, 512);

  attn_wmma<<<dim3(32, 256), 256, 0, stream>>>(qbuf, kbuf, vbuf, obuf);

  gemm_bigtile<false, false><<<dim3(4, 1024), 256, 0, stream>>>(obuf, WwT, bw,
                                                                out, 65536, 1024);
}
